// GraphMLPMixer4TreeNeighbour_80659485818911
// MI455X (gfx1250) — matrix operand order repaired and checked by reading, not compiled, not run on hardware
//
#include <hip/hip_runtime.h>
#include <hip/hip_bf16.h>

// ---------------------------------------------------------------------------
// GraphMLPMixer forward for MI455X (gfx1250, wave32, WMMA).
//
// Roofline: ~150 GFLOP of GEMM vs ~3-4 GB of gather/scatter traffic.
// At 23.3 TB/s HBM the irregular passes dominate (~150-200us), so GEMMs run
// on v_wmma_f32_16x16x32_bf16 (bf16 in, f32 accum) and are ~free. Weights are
// pre-packed into WMMA-fragment order so the GEMM inner loop is pure
// global_load_b128 + v_wmma (no scalar u16 gathers). Edge features live in
// bf16 to halve their 4x-per-model read cost; the ReLU-gated message scatter
// skips zero-valued atomics.
// ---------------------------------------------------------------------------

typedef __attribute__((ext_vector_type(16))) __bf16 v16bf;
typedef __attribute__((ext_vector_type(8)))  __bf16 v8bf;
typedef __attribute__((ext_vector_type(8)))  float  v8f;

#define D_      128
#define N_      100000
#define NS_     400000
#define E_      200000
#define ES_     800000
#define B_      128
#define P_      32
#define BP_     (B_ * P_)
#define NLG_    4
#define NLM_    4
#define NOUT_   32
#define NROOT_  128
#define TOKH_   64
#define CHH_    512

#define GS_LOOP(i, n) for (long long i = (long long)blockIdx.x * blockDim.x + threadIdx.x; \
                           i < (long long)(n); i += (long long)gridDim.x * blockDim.x)

static inline unsigned ew_blocks(long long n) {
  long long b = (n + 255) / 256;
  if (b < 1) b = 1;
  if (b > 524288) b = 524288;
  return (unsigned)b;
}

// --------------------------- WMMA GEMM -------------------------------------
// C[M,N] = A[M,K](bf16 row-major) @ B(packed fragments) (+ bias[N]).
// One wave per 16x16 tile. A fragment = two contiguous b128 loads per lane;
// B is pre-packed so its fragment is two contiguous b128 loads per lane.
// KT is compile-time -> fully unrolled K loop (KT/32 wmma per wave).
template <int KT>
__global__ void k_wmma_gemm(const __bf16* __restrict__ A, const __bf16* __restrict__ Bp,
                            const float* __restrict__ bias, float* __restrict__ C,
                            int M, int N) {
  int wave = blockIdx.x * (blockDim.x >> 5) + ((int)threadIdx.x >> 5);
  int lane = threadIdx.x & 31;
  int mt = M >> 4, nt = N >> 4;
  if (wave >= mt * nt) return;                 // uniform per-wave: EXEC stays all-ones
  int mtile = wave / nt, ntile = wave % nt;

  int m = mtile * 16 + (lane & 15);
  int khalf = lane >> 4;
  const v8bf* __restrict__ A8 = (const v8bf*)A;
  // packed B: [ntile][kk/32][lane][16] contiguous bf16
  const __bf16* __restrict__ Bfrag = Bp + ((long)ntile * (KT / 32) * 32 + lane) * 16;

  v8f acc = {};
#pragma unroll
  for (int kk = 0; kk < KT; kk += 32) {
    if (kk + 32 < KT)  // prefetch next A chunk (global_prefetch_b8, near-temporal)
      __builtin_prefetch(A + (long)m * KT + kk + 32, 0, 3);
    long abase = (long)m * KT + kk + khalf * 16;         // multiple of 8 elems (16B)
    v8bf alo = A8[abase >> 3];
    v8bf ahi = A8[(abase + 8) >> 3];
    v16bf a = __builtin_shufflevector(alo, ahi, 0, 1, 2, 3, 4, 5, 6, 7,
                                      8, 9, 10, 11, 12, 13, 14, 15);
    const v8bf* B8 = (const v8bf*)(Bfrag + (long)(kk >> 5) * 32 * 16);
    v8bf blo = B8[0];
    v8bf bhi = B8[1];
    v16bf b = __builtin_shufflevector(blo, bhi, 0, 1, 2, 3, 4, 5, 6, 7,
                                      8, 9, 10, 11, 12, 13, 14, 15);
    acc = __builtin_amdgcn_wmma_f32_16x16x32_bf16(false, a, false, b, (short)0, acc,
                                                  false, false);
  }
  int n = ntile * 16 + (lane & 15);
  int mbase = mtile * 16 + (lane >> 4) * 8;    // ISA f32 C/D layout
  float bv = bias ? bias[n] : 0.0f;
#pragma unroll
  for (int r = 0; r < 8; ++r)
    C[(long)(mbase + r) * N + n] = acc[r] + bv;
}

// Pre-pack f32 weight [K,N] row-major -> bf16 WMMA B-fragment order:
// out[(((ntile*(K/32) + kk32)*32 + lane)*16 + j] = W[(kk32*32 + (lane>>4)*16 + j)*N
//                                                    + ntile*16 + (lane&15)]
__global__ void k_pack_b(const float* __restrict__ Wf, __bf16* __restrict__ out,
                         int K, int N) {
  long long tot = (long long)K * N;
  GS_LOOP(i, tot) {
    int j = (int)(i & 15);
    long long t = i >> 4;
    int lane = (int)(t & 31); t >>= 5;
    int kk32 = (int)(t % (K / 32));
    int ntile = (int)(t / (K / 32));
    int n = ntile * 16 + (lane & 15);
    int k = kk32 * 32 + (lane >> 4) * 16 + j;
    out[i] = (__bf16)Wf[(long)k * N + n];
  }
}

// --------------------------- elementwise / graph kernels --------------------
__global__ void k_embed(const int* __restrict__ xkv, const float* __restrict__ ke,
                        const float* __restrict__ ve, float* __restrict__ h) {
  GS_LOOP(i, (long long)N_ * D_) {
    long long r = i >> 7; int d = (int)(i & 127);
    h[i] = ke[(long)xkv[2 * r] * D_ + d] + ve[(long)xkv[2 * r + 1] * D_ + d];
  }
}

__global__ void k_gather_rows(const float* __restrict__ in, const int* __restrict__ map,
                              float* __restrict__ out, long long rows, int C) {
  GS_LOOP(i, rows * C) {
    long long r = i / C; int c = (int)(i % C);
    out[i] = in[(long)map[r] * C + c];
  }
}

__global__ void k_gather_edge_bf16(const int* __restrict__ ea, const int* __restrict__ emap,
                                   const float* __restrict__ ee, __bf16* __restrict__ es) {
  GS_LOOP(i, (long long)ES_ * D_) {
    long long r = i >> 7; int d = (int)(i & 127);
    es[i] = (__bf16)ee[(long)ea[emap[r]] * D_ + d];
  }
}

__global__ void k_seg_add(const float* __restrict__ x, const int* __restrict__ seg,
                          float* __restrict__ sum, long long rows) {
  GS_LOOP(i, rows * D_) {
    long long r = i >> 7; int d = (int)(i & 127);
    atomicAdd(&sum[(long)seg[r] * D_ + d], x[i]);
  }
}

__global__ void k_seg_count(const int* __restrict__ seg, float* __restrict__ cnt,
                            long long rows) {
  GS_LOOP(i, rows) atomicAdd(&cnt[seg[i]], 1.0f);
}

__global__ void k_div_cnt(float* __restrict__ acc, const float* __restrict__ cnt,
                          long long S, int C) {
  GS_LOOP(i, S * C) acc[i] /= fmaxf(cnt[i / C], 1.0f);
}

// GINE message: agg[dst] += relu(h[src] + e); skip zero atomics.
__global__ void k_gine_scatter(const float* __restrict__ h, const __bf16* __restrict__ e,
                               const int* __restrict__ src, const int* __restrict__ dst,
                               float* __restrict__ agg) {
  GS_LOOP(i, (long long)ES_ * D_) {
    long long ed = i >> 7; int d = (int)(i & 127);
    float v = h[(long)src[ed] * D_ + d] + (float)e[i];
    if (v > 0.0f) atomicAdd(&agg[(long)dst[ed] * D_ + d], v);
  }
}

__global__ void k_add_bf16(const float* __restrict__ a, const float* __restrict__ b,
                           __bf16* __restrict__ out, long long n) {
  GS_LOOP(i, n) out[i] = (__bf16)(a[i] + b[i]);
}

__global__ void k_f32_to_bf16(const float* __restrict__ in, __bf16* __restrict__ out,
                              long long n) {
  GS_LOOP(i, n) out[i] = (__bf16)in[i];
}

// Column sums / sum-of-squares for BatchNorm; optional row gather (seg != null).
__global__ void k_colstats(const float* __restrict__ X, const int* __restrict__ seg,
                           long long rows, int C, int rpb, float* __restrict__ stats) {
  int c = threadIdx.x;                         // blockDim.x == C
  long long r0 = (long long)blockIdx.x * rpb;
  long long r1 = r0 + rpb; if (r1 > rows) r1 = rows;
  float s = 0.f, q = 0.f;
  for (long long r = r0; r < r1; ++r) {
    long long row = seg ? (long long)seg[r] : r;
    float v = X[row * C + c];
    s += v; q += v * v;
  }
  atomicAdd(&stats[c], s);
  atomicAdd(&stats[C + c], q);
}

__device__ inline float bn_val(float x, const float* stats, int C, int c, float invM,
                               const float* g, const float* b) {
  float m = stats[c] * invM;
  float var = fmaxf(stats[C + c] * invM - m * m, 0.f);
  return (x - m) * rsqrtf(var + 1e-5f) * g[c] + b[c];
}

__global__ void k_bn_relu_bf16(const float* __restrict__ x, const float* __restrict__ stats,
                               float invM, const float* __restrict__ g,
                               const float* __restrict__ b, __bf16* __restrict__ out,
                               long long rows, int C) {
  GS_LOOP(i, rows * C) {
    int c = (int)(i % C);
    out[i] = (__bf16)fmaxf(bn_val(x[i], stats, C, c, invM, g, b), 0.f);
  }
}

__global__ void k_bn_relu_res(const float* __restrict__ x, const float* __restrict__ stats,
                              float invM, const float* __restrict__ g,
                              const float* __restrict__ b, float* __restrict__ h,
                              long long rows) {
  GS_LOOP(i, rows * D_) {
    int c = (int)(i & 127);
    h[i] += fmaxf(bn_val(x[i], stats, D_, c, invM, g, b), 0.f);
  }
}

__global__ void k_bn_relu_gather_add(const float* __restrict__ z, const float* __restrict__ stats,
                                     float invM, const float* __restrict__ g,
                                     const float* __restrict__ b, const int* __restrict__ seg,
                                     float* __restrict__ h, long long rows) {
  GS_LOOP(i, rows * D_) {
    long long r = i >> 7; int c = (int)(i & 127);
    float v = z[(long)seg[r] * D_ + c];
    h[i] += fmaxf(bn_val(v, stats, D_, c, invM, g, b), 0.f);
  }
}

// LayerNorm over D=128 per row; optional f32 / bf16 / transposed-bf16 outputs.
__global__ void k_layernorm(const float* __restrict__ X, const float* __restrict__ g,
                            const float* __restrict__ b, float* out_f32,
                            __bf16* out_b16, __bf16* outT, int P) {
  long long row = blockIdx.x;
  int d = threadIdx.x;
  float x = X[row * D_ + d];
  __shared__ float sh[D_];
  sh[d] = x; __syncthreads();
#pragma unroll
  for (int o = 64; o > 0; o >>= 1) { if (d < o) sh[d] += sh[d + o]; __syncthreads(); }
  float mean = sh[0] * (1.0f / D_);
  __syncthreads();
  float dx = x - mean;
  sh[d] = dx * dx; __syncthreads();
#pragma unroll
  for (int o = 64; o > 0; o >>= 1) { if (d < o) sh[d] += sh[d + o]; __syncthreads(); }
  float var = sh[0] * (1.0f / D_);
  float y = dx * rsqrtf(var + 1e-5f) * g[d] + b[d];
  if (out_f32) out_f32[row * D_ + d] = y;
  if (out_b16) out_b16[row * D_ + d] = (__bf16)y;
  if (outT) {
    long long bg = row / P; int p = (int)(row % P);
    outT[(bg * D_ + d) * (long long)P + p] = (__bf16)y;
  }
}

__global__ void k_gelu_bf16(const float* __restrict__ in, __bf16* __restrict__ out,
                            long long n) {
  GS_LOOP(i, n) {
    float x = in[i];
    float t = tanhf(0.7978845608028654f * (x + 0.044715f * x * x * x));
    out[i] = (__bf16)(0.5f * x * (1.0f + t));
  }
}

__global__ void k_add_token(float* __restrict__ mx, const float* __restrict__ yT,
                            long long n) {
  GS_LOOP(i, n) {
    long long row = i >> 7; int d = (int)(i & 127);
    long long bg = row >> 5; int p = (int)(row & 31);
    mx[i] += yT[(bg * D_ + d) * (long long)P_ + p];
  }
}

__global__ void k_add_inplace(float* __restrict__ a, const float* __restrict__ b,
                              long long n) {
  GS_LOOP(i, n) a[i] += b[i];
}

__global__ void k_concat_scatter(const float* __restrict__ h, const float* __restrict__ mixx,
                                 const int* __restrict__ nmap, const int* __restrict__ batch,
                                 float* __restrict__ acc) {
  GS_LOOP(i, (long long)NS_ * D_) {
    long long r = i >> 7; int d = (int)(i & 127);
    long long nidx = nmap[r];
    atomicAdd(&acc[nidx * (2 * D_) + d], h[i]);
    atomicAdd(&acc[nidx * (2 * D_) + D_ + d], mixx[(long)batch[r] * D_ + d]);
  }
}

__global__ void k_gather_root_bf16(const float* __restrict__ acc, const int* __restrict__ ridx,
                                   __bf16* __restrict__ out) {
  GS_LOOP(i, (long long)NROOT_ * (2 * D_)) {
    long long r = i / (2 * D_); int c = (int)(i % (2 * D_));
    out[i] = (__bf16)acc[(long)ridx[r] * (2 * D_) + c];
  }
}

// --------------------------- host orchestration ----------------------------
static void gemm(hipStream_t s, const __bf16* A, const __bf16* Bp, const float* bias,
                 float* C, int M, int N, int K) {
  long long tiles = (long long)(M / 16) * (N / 16);
  unsigned blocks = (unsigned)((tiles + 7) / 8);
  switch (K) {
    case 32:  k_wmma_gemm<32 ><<<blocks, 256, 0, s>>>(A, Bp, bias, C, M, N); break;
    case 64:  k_wmma_gemm<64 ><<<blocks, 256, 0, s>>>(A, Bp, bias, C, M, N); break;
    case 128: k_wmma_gemm<128><<<blocks, 256, 0, s>>>(A, Bp, bias, C, M, N); break;
    case 256: k_wmma_gemm<256><<<blocks, 256, 0, s>>>(A, Bp, bias, C, M, N); break;
    case 512: k_wmma_gemm<512><<<blocks, 256, 0, s>>>(A, Bp, bias, C, M, N); break;
    default: break;
  }
}

template <typename T>
static T* carve(char* base, size_t* off, size_t count) {
  T* p = (T*)(base + *off);
  *off = (*off + count * sizeof(T) + 255) & ~(size_t)255;
  return p;
}

extern "C" void kernel_launch(void* const* d_in, const int* in_sizes, int n_in,
                              void* d_out, int out_size, void* d_ws, size_t ws_size,
                              hipStream_t stream) {
  // ---- parameter pytree (JAX sorted-key flatten order), 91 leaves ----
  static const long LSZ[91] = {
    128,128,16384, 128,128,16384, 128,128,16384,          // U x3: b,g,w
    32, 256,256, 65536, 8192,                             // dec_b2, dec_bn_b, dec_bn_g, dec_w1, dec_w2
    1024,                                                 // edge_emb
    128,128,128,128,16384,16384, 128,128,128,128,16384,16384, // gnn x4: bn1_b,bn1_g,bn_b,bn_g,w1,w2
    128,128,128,128,16384,16384, 128,128,128,128,16384,16384,
    65536, 128,128,                                       // key_emb, ln_f_b, ln_f_g
    512,128,65536,65536,128,128,128,128,64,32,2048,2048,  // mixer x4
    512,128,65536,65536,128,128,128,128,64,32,2048,2048,
    512,128,65536,65536,128,128,128,128,64,32,2048,2048,
    512,128,65536,65536,128,128,128,128,64,32,2048,2048,
    65536                                                 // val_emb
  };
  const float* L[91];
  int ib;  // index of first non-param input
  if (n_in >= 98) { for (int i = 0; i < 91; ++i) L[i] = (const float*)d_in[i]; ib = 91; }
  else { const float* base = (const float*)d_in[0]; long off = 0;
         for (int i = 0; i < 91; ++i) { L[i] = base + off; off += LSZ[i]; } ib = 1; }

  const float* dec_b2 = L[9];  const float* dec_bn_b = L[10]; const float* dec_bn_g = L[11];
  const float* dec_w1 = L[12]; const float* dec_w2 = L[13];   const float* edge_emb = L[14];
  const float* key_emb = L[39]; const float* ln_f_b = L[40];  const float* ln_f_g = L[41];
  const float* val_emb = L[90];
#define UL(i, j)   L[3 * (i) + (j)]            // j: 0=b 1=g 2=w
#define GL(i, j)   L[15 + 6 * (i) + (j)]       // j: 0=bn1_b 1=bn1_g 2=bn_b 3=bn_g 4=w1 5=w2
#define ML(i, j)   L[42 + 12 * (i) + (j)]      // j: 0=ch_b1 1=ch_b2 2=ch_w1 3=ch_w2 4=ln1_b
                                               //    5=ln1_g 6=ln2_b 7=ln2_g 8=tok_b1 9=tok_b2 10=tok_w1 11=tok_w2
  const int* x_nodes  = (const int*)d_in[ib + 0];
  const int* edge_att = (const int*)d_in[ib + 1];
  const int* nmap     = (const int*)d_in[ib + 2];
  const int* emap     = (const int*)d_in[ib + 3];
  const int* comb     = (const int*)d_in[ib + 4];
  const int* batch    = (const int*)d_in[ib + 5];
  const int* ridx     = (const int*)d_in[ib + 6];
  const int* src = comb;
  const int* dst = comb + ES_;
  (void)in_sizes; (void)ws_size; (void)out_size;

  // ---- workspace carve ----
  char* W = (char*)d_ws; size_t o = 0;
  float*  h_node  = carve<float >(W, &o, (size_t)N_ * D_);
  float*  h_s     = carve<float >(W, &o, (size_t)NS_ * D_);
  float*  scr     = carve<float >(W, &o, (size_t)NS_ * D_);     // agg / t1 / t2
  __bf16* e_s     = carve<__bf16>(W, &o, (size_t)ES_ * D_);
  __bf16* bf1     = carve<__bf16>(W, &o, (size_t)NS_ * D_);
  __bf16* bf2     = carve<__bf16>(W, &o, (size_t)NS_ * D_);
  float*  meanBP  = carve<float >(W, &o, (size_t)BP_ * D_);
  float*  cntBP   = carve<float >(W, &o, BP_);
  float*  zBP     = carve<float >(W, &o, (size_t)BP_ * D_);
  __bf16* subxb   = carve<__bf16>(W, &o, (size_t)BP_ * D_);
  float*  meanN   = carve<float >(W, &o, (size_t)N_ * D_);
  float*  cntN    = carve<float >(W, &o, N_);
  float*  accN2   = carve<float >(W, &o, (size_t)N_ * 2 * D_);
  float*  stats   = carve<float >(W, &o, 2 * 256);
  __bf16* tTb     = carve<__bf16>(W, &o, (size_t)B_ * D_ * P_);
  float*  tok1    = carve<float >(W, &o, (size_t)B_ * D_ * TOKH_);
  __bf16* tok1b   = carve<__bf16>(W, &o, (size_t)B_ * D_ * TOKH_);
  float*  tok2    = carve<float >(W, &o, (size_t)B_ * D_ * P_);
  __bf16* lnb     = carve<__bf16>(W, &o, (size_t)BP_ * D_);
  float*  ch1     = carve<float >(W, &o, (size_t)BP_ * CHH_);
  __bf16* ch1b    = carve<__bf16>(W, &o, (size_t)BP_ * CHH_);
  float*  ch2     = carve<float >(W, &o, (size_t)BP_ * D_);
  float*  mixer_x = carve<float >(W, &o, (size_t)BP_ * D_);
  __bf16* rootb   = carve<__bf16>(W, &o, (size_t)NROOT_ * 2 * D_);
  float*  decz    = carve<float >(W, &o, (size_t)NROOT_ * 2 * D_);
  __bf16* deczb   = carve<__bf16>(W, &o, (size_t)NROOT_ * 2 * D_);
  __bf16* wg1[NLG_]; __bf16* wg2[NLG_]; __bf16* wU[NLG_ - 1];
  __bf16* wt1[NLM_]; __bf16* wt2[NLM_]; __bf16* wc1[NLM_]; __bf16* wc2[NLM_];
  for (int i = 0; i < NLG_; ++i) { wg1[i] = carve<__bf16>(W, &o, 16384); wg2[i] = carve<__bf16>(W, &o, 16384); }
  for (int i = 0; i < NLG_ - 1; ++i) wU[i] = carve<__bf16>(W, &o, 16384);
  for (int i = 0; i < NLM_; ++i) { wt1[i] = carve<__bf16>(W, &o, 2048);  wt2[i] = carve<__bf16>(W, &o, 2048);
                                   wc1[i] = carve<__bf16>(W, &o, 65536); wc2[i] = carve<__bf16>(W, &o, 65536); }
  __bf16* wd1 = carve<__bf16>(W, &o, 65536);
  __bf16* wd2 = carve<__bf16>(W, &o, 8192);

#define EW(kern, n, ...) kern<<<ew_blocks((long long)(n)), 256, 0, stream>>>(__VA_ARGS__)
#define ZERO(p, elems) hipMemsetAsync((p), 0, (size_t)(elems) * sizeof(float), stream)

  // ---- weight pre-pack (f32 -> bf16 WMMA fragment order) ----
  for (int i = 0; i < NLG_; ++i) { EW(k_pack_b, 16384, GL(i,4), wg1[i], 128, 128);
                                   EW(k_pack_b, 16384, GL(i,5), wg2[i], 128, 128); }
  for (int i = 0; i < NLG_ - 1; ++i) EW(k_pack_b, 16384, UL(i,2), wU[i], 128, 128);
  for (int i = 0; i < NLM_; ++i) { EW(k_pack_b, 2048,  ML(i,10), wt1[i], 32, 64);
                                   EW(k_pack_b, 2048,  ML(i,11), wt2[i], 64, 32);
                                   EW(k_pack_b, 65536, ML(i,2),  wc1[i], 128, 512);
                                   EW(k_pack_b, 65536, ML(i,3),  wc2[i], 512, 128); }
  EW(k_pack_b, 65536, dec_w1, wd1, 256, 256);
  EW(k_pack_b, 8192,  dec_w2, wd2, 256, 32);

  // ---- segment counts (reused) ----
  ZERO(cntBP, BP_); ZERO(cntN, N_);
  EW(k_seg_count, NS_, batch, cntBP, NS_);
  EW(k_seg_count, NS_, nmap,  cntN,  NS_);

  // ---- embeddings + gathers ----
  EW(k_embed, (long long)N_ * D_, x_nodes, key_emb, val_emb, h_node);
  EW(k_gather_rows, (long long)NS_ * D_, h_node, nmap, h_s, NS_, D_);
  EW(k_gather_edge_bf16, (long long)ES_ * D_, edge_att, emap, edge_emb, e_s);

  // ---- GNN layers ----
  for (int i = 0; i < NLG_; ++i) {
    if (i > 0) {
      // sub = seg_mean(h, batch)[batch]; h += relu(bn(sub @ Uw)); h = seg_mean(h, nmap)[nmap]
      ZERO(meanBP, (size_t)BP_ * D_);
      EW(k_seg_add, (long long)NS_ * D_, h_s, batch, meanBP, NS_);
      EW(k_div_cnt, (long long)BP_ * D_, meanBP, cntBP, BP_, D_);
      EW(k_f32_to_bf16, (long long)BP_ * D_, meanBP, subxb, (long long)BP_ * D_);
      gemm(stream, subxb, wU[i - 1], nullptr, zBP, BP_, D_, D_);
      ZERO(stats, 2 * D_);
      k_colstats<<<(NS_ + 1023) / 1024, D_, 0, stream>>>(zBP, batch, NS_, D_, 1024, stats);
      EW(k_bn_relu_gather_add, (long long)NS_ * D_, zBP, stats, 1.0f / NS_,
         UL(i - 1, 1), UL(i - 1, 0), batch, h_s, NS_);
      ZERO(meanN, (size_t)N_ * D_);
      EW(k_seg_add, (long long)NS_ * D_, h_s, nmap, meanN, NS_);
      EW(k_div_cnt, (long long)N_ * D_, meanN, cntN, N_, D_);
      EW(k_gather_rows, (long long)NS_ * D_, meanN, nmap, h_s, NS_, D_);
    }
    // GINE: agg = scatter_sum(relu(h[src]+e)); t1 = relu(bn1((h+agg)@w1)); t2 = t1@w2
    ZERO(scr, (size_t)NS_ * D_);
    EW(k_gine_scatter, (long long)ES_ * D_, h_s, e_s, src, dst, scr);
    EW(k_add_bf16, (long long)NS_ * D_, h_s, scr, bf1, (long long)NS_ * D_);
    gemm(stream, bf1, wg1[i], nullptr, scr, NS_, D_, D_);
    ZERO(stats, 2 * D_);
    k_colstats<<<(NS_ + 1023) / 1024, D_, 0, stream>>>(scr, nullptr, NS_, D_, 1024, stats);
    EW(k_bn_relu_bf16, (long long)NS_ * D_, scr, stats, 1.0f / NS_, GL(i,1), GL(i,0),
       bf2, NS_, D_);
    gemm(stream, bf2, wg2[i], nullptr, scr, NS_, D_, D_);
    // h = relu(bn(t2)) + h
    ZERO(stats, 2 * D_);
    k_colstats<<<(NS_ + 1023) / 1024, D_, 0, stream>>>(scr, nullptr, NS_, D_, 1024, stats);
    EW(k_bn_relu_res, (long long)NS_ * D_, scr, stats, 1.0f / NS_, GL(i,3), GL(i,2), h_s, NS_);
  }

  // ---- patch pooling -> mixer input mx = meanBP [B,P,D] ----
  ZERO(meanBP, (size_t)BP_ * D_);
  EW(k_seg_add, (long long)NS_ * D_, h_s, batch, meanBP, NS_);
  EW(k_div_cnt, (long long)BP_ * D_, meanBP, cntBP, BP_, D_);
  float* mx = meanBP;

  // ---- MLP-Mixer ----
  for (int l = 0; l < NLM_; ++l) {
    // token mixing (operates on [B*D, P])
    k_layernorm<<<BP_, D_, 0, stream>>>(mx, ML(l,5), ML(l,4), nullptr, nullptr, tTb, P_);
    gemm(stream, tTb, wt1[l], ML(l,8), tok1, B_ * D_, TOKH_, P_);
    EW(k_gelu_bf16, (long long)B_ * D_ * TOKH_, tok1, tok1b, (long long)B_ * D_ * TOKH_);
    gemm(stream, tok1b, wt2[l], ML(l,9), tok2, B_ * D_, P_, TOKH_);
    EW(k_add_token, (long long)BP_ * D_, mx, tok2, (long long)BP_ * D_);
    // channel mixing ([B*P, D])
    k_layernorm<<<BP_, D_, 0, stream>>>(mx, ML(l,7), ML(l,6), nullptr, lnb, nullptr, P_);
    gemm(stream, lnb, wc1[l], ML(l,0), ch1, BP_, CHH_, D_);
    EW(k_gelu_bf16, (long long)BP_ * CHH_, ch1, ch1b, (long long)BP_ * CHH_);
    gemm(stream, ch1b, wc2[l], ML(l,1), ch2, BP_, D_, CHH_);
    EW(k_add_inplace, (long long)BP_ * D_, mx, ch2, (long long)BP_ * D_);
  }
  k_layernorm<<<BP_, D_, 0, stream>>>(mx, ln_f_g, ln_f_b, mixer_x, nullptr, nullptr, P_);

  // ---- concat + seg_mean over nodes -> [N, 2D] ----
  ZERO(accN2, (size_t)N_ * 2 * D_);
  EW(k_concat_scatter, (long long)NS_ * D_, h_s, mixer_x, nmap, batch, accN2);
  EW(k_div_cnt, (long long)N_ * 2 * D_, accN2, cntN, N_, 2 * D_);

  // ---- decoder ----
  EW(k_gather_root_bf16, (long long)NROOT_ * 2 * D_, accN2, ridx, rootb);
  gemm(stream, rootb, wd1, nullptr, decz, NROOT_, 2 * D_, 2 * D_);
  ZERO(stats, 2 * 256);
  k_colstats<<<1, 2 * D_, 0, stream>>>(decz, nullptr, NROOT_, 2 * D_, 1024, stats);
  EW(k_bn_relu_bf16, (long long)NROOT_ * 2 * D_, decz, stats, 1.0f / NROOT_,
     dec_bn_g, dec_bn_b, deczb, NROOT_, 2 * D_);
  gemm(stream, deczb, wd2, dec_b2, (float*)d_out, NROOT_, NOUT_, 2 * D_);
}
